// PatchedDensityNetwork_14611478741019
// MI455X (gfx1250) — matrix operand order repaired and checked by reading, not compiled, not run on hardware
//
#include <hip/hip_runtime.h>

// ---------- types for WMMA ----------
typedef __bf16 bf16_t;
typedef bf16_t v16bf __attribute__((ext_vector_type(16)));
typedef bf16_t v8bf  __attribute__((ext_vector_type(8)));
typedef float  v8f   __attribute__((ext_vector_type(8)));

union Frag16 { v16bf v; v8bf h[2]; };

// ---------- helpers ----------
__device__ __forceinline__ unsigned short f32_to_bf16_rne(float f) {
  union { float f; unsigned u; } c; c.f = f;
  unsigned u = c.u;
  if ((u & 0x7F800000u) == 0x7F800000u) return (unsigned short)(u >> 16); // Inf/NaN passthrough
  unsigned r = u + 0x7FFFu + ((u >> 16) & 1u); // round to nearest even
  return (unsigned short)(r >> 16);
}

// out[i] = bf16(W[i] * mask[i])   (mask may be null)
__global__ void cvt_masked_bf16(const float* __restrict__ W,
                                const float* __restrict__ mask,
                                unsigned short* __restrict__ out, int n) {
  int i = blockIdx.x * blockDim.x + threadIdx.x;
  if (i >= n) return;
  float v = W[i];
  if (mask) v *= mask[i];
  out[i] = f32_to_bf16_rne(v);
}

// out[i] = bf16(relu(a[i]))
__global__ void relu_cvt_bf16(const float* __restrict__ a,
                              unsigned short* __restrict__ out, int n) {
  int i = blockIdx.x * blockDim.x + threadIdx.x;
  if (i >= n) return;
  float v = a[i];
  v = v > 0.f ? v : 0.f;
  out[i] = f32_to_bf16_rne(v);
}

// ---------- bf16 WMMA GEMM:  C[M,N] = A[M,K] * Wt[N,K]^T + bias[N] ----------
// A, Wt row-major bf16 (stored as u16). C row-major f32.
// Optional second output: Obf[i] = bf16(relu(C[i])) fused (layer-0 fast path).
// grid = (N/256, M/128), block = 256 (8 waves).
// Waves: wm = wave&1 (2 rows of 64), wn = wave>>1 (4 cols of 64).
// Wave tile 64x64 = 4x4 WMMA tiles -> 16 WMMAs per 16 b128 loads per K=32 step.
__global__ __launch_bounds__(256) void gemm_bf16_wmma(
    const unsigned short* __restrict__ A,
    const unsigned short* __restrict__ Wt,
    const float* __restrict__ bias,
    float* __restrict__ C,
    unsigned short* __restrict__ Obf,   // nullable: bf16(relu(C)) fused output
    int M, int N, int K) {
  const int tid  = threadIdx.x;
  const int wave = tid >> 5;
  const int lane = tid & 31;
  const int kh   = lane >> 4;   // lane half selects K sub-range per ISA fragment layout
  const int l16  = lane & 15;

  const int wm = wave & 1;      // 2 wave-rows
  const int wn = wave >> 1;     // 4 wave-cols
  const int m_base = blockIdx.y * 128 + wm * 64;
  const int n_base = blockIdx.x * 256 + wn * 64;

  v8f acc[4][4] = {};           // f32 accumulators (8 VGPRs per 16x16 tile)

  // A 16-bit 16x32 fragment: lane l16 = row M; per lane two contiguous 8-elem
  // chunks at K = kh*8 and K = 16 + kh*8.
  const unsigned short* Abase = A  + (size_t)(m_base + l16) * K + kh * 8;
  // B 16-bit 32x16 fragment: lane l16 = col N; per lane two contiguous 8-elem
  // chunks at K = kh*8 and K = 16 + kh*8.
  const unsigned short* Bbase = Wt + (size_t)(n_base + l16) * K + kh * 8;

  for (int k = 0; k < K; k += 32) {
    Frag16 af[4];
#pragma unroll
    for (int mt = 0; mt < 4; ++mt) {
      const unsigned short* ap = Abase + (size_t)(mt * 16) * K + k;
      af[mt].h[0] = *(const v8bf*)(ap);
      af[mt].h[1] = *(const v8bf*)(ap + 16);
    }
    Frag16 bw[4];
#pragma unroll
    for (int nt = 0; nt < 4; ++nt) {
      const unsigned short* bp = Bbase + (size_t)(nt * 16) * K + k;
      bw[nt].h[0] = *(const v8bf*)(bp);
      bw[nt].h[1] = *(const v8bf*)(bp + 16);
    }
#pragma unroll
    for (int mt = 0; mt < 4; ++mt)
#pragma unroll
      for (int nt = 0; nt < 4; ++nt)
        acc[mt][nt] = __builtin_amdgcn_wmma_f32_16x16x32_bf16(
            false, af[mt].v, false, bw[nt].v, (short)0, acc[mt][nt],
            false, false);
  }

  // C/D 16x16 f32 layout: VGPR r, lanes 0-15 -> M=r, lanes 16-31 -> M=8+r; N=l16.
#pragma unroll
  for (int nt = 0; nt < 4; ++nt) {
    const int col = n_base + nt * 16 + l16;
    const float bv = bias[col];
#pragma unroll
    for (int mt = 0; mt < 4; ++mt) {
#pragma unroll
      for (int r = 0; r < 8; ++r) {
        const int row = m_base + mt * 16 + kh * 8 + r;
        const float v = acc[mt][nt][r] + bv;
        C[(size_t)row * N + col] = v;
        if (Obf) Obf[(size_t)row * N + col] = f32_to_bf16_rne(v > 0.f ? v : 0.f);
      }
    }
  }
}

// ---------- patch networks (tiny, one thread per batch row) ----------
// acts_prev: B x Kprev pre-ReLU activations of previous layer.
// out:       B x Nout, already holds sparse_out (+bias); we accumulate patch.
__global__ void patch_kernel(const float* __restrict__ acts_prev,
                             float* __restrict__ out, int Nout,
                             const int* __restrict__ hsrc, const int* __restrict__ htgt,
                             const int* __restrict__ lsmp, const int* __restrict__ ltgt,
                             const float* __restrict__ hW1, const float* __restrict__ hb1,
                             const float* __restrict__ hW2, const float* __restrict__ hb2,
                             const float* __restrict__ lW1, const float* __restrict__ lb1,
                             const float* __restrict__ lW2, const float* __restrict__ lb2,
                             int Bsz, int Kprev) {
  int b = blockIdx.x * blockDim.x + threadIdx.x;
  if (b >= Bsz) return;
  const float* arow = acts_prev + (size_t)b * Kprev;
  float* orow = out + (size_t)b * Nout;
#pragma unroll
  for (int p = 0; p < 5; ++p) {
    // h-patch: scalar col -> 8 hidden -> 4 outputs
    float col = arow[hsrc[p]];
    float ho[4];
#pragma unroll
    for (int o = 0; o < 4; ++o) ho[o] = hb2[p * 4 + o];
#pragma unroll
    for (int h = 0; h < 8; ++h) {
      float hid = col * hW1[p * 8 + h] + hb1[p * 8 + h];
      hid = hid > 0.f ? hid : 0.f;
#pragma unroll
      for (int o = 0; o < 4; ++o) ho[o] += hid * hW2[(p * 4 + o) * 8 + h];
    }
#pragma unroll
    for (int o = 0; o < 4; ++o) orow[htgt[p * 4 + o]] += 0.3f * ho[o];

    // l-patch: 20 relu'd samples -> 8 hidden -> 1 output
    float g[20];
#pragma unroll
    for (int i = 0; i < 20; ++i) {
      float v = arow[lsmp[p * 20 + i]];
      g[i] = v > 0.f ? v : 0.f;
    }
    float lo = lb2[p];
#pragma unroll
    for (int h = 0; h < 8; ++h) {
      float s = lb1[p * 8 + h];
#pragma unroll
      for (int i = 0; i < 20; ++i) s += g[i] * lW1[(p * 20 + i) * 8 + h];
      s = s > 0.f ? s : 0.f;
      lo += s * lW2[p * 8 + h];
    }
    orow[ltgt[p]] += 0.5f * lo;
  }
}

// ---------- launcher ----------
extern "C" void kernel_launch(void* const* d_in, const int* in_sizes, int n_in,
                              void* d_out, int out_size, void* d_ws, size_t ws_size,
                              hipStream_t stream) {
  (void)in_sizes; (void)n_in; (void)out_size; (void)ws_size;
  const float* x     = (const float*)d_in[0];
  const float* W0    = (const float*)d_in[1];
  const float* b0    = (const float*)d_in[2];
  const float* mask0 = (const float*)d_in[3];
  const float* W1    = (const float*)d_in[4];
  const float* b1    = (const float*)d_in[5];
  const float* mask1 = (const float*)d_in[6];
  const float* W2    = (const float*)d_in[7];
  const float* b2    = (const float*)d_in[8];
  const float* mask2 = (const float*)d_in[9];
  const float* hW1   = (const float*)d_in[10];
  const float* hb1   = (const float*)d_in[11];
  const float* hW2   = (const float*)d_in[12];
  const float* hb2   = (const float*)d_in[13];
  const float* lW1   = (const float*)d_in[14];
  const float* lb1   = (const float*)d_in[15];
  const float* lW2   = (const float*)d_in[16];
  const float* lb2   = (const float*)d_in[17];
  const int*   hsrc  = (const int*)d_in[18];
  const int*   htgt  = (const int*)d_in[19];
  const int*   lsmp  = (const int*)d_in[20];
  const int*   ltgt  = (const int*)d_in[21];

  constexpr int Bsz = 8192, D0 = 3072, D1 = 4096, D2 = 4096, D3 = 1024, P = 5;

  // workspace bump allocator
  unsigned char* wsb = (unsigned char*)d_ws;
  size_t off = 0;
  auto take = [&](size_t bytes) -> void* {
    void* p = wsb + off;
    off = (off + bytes + 255) & ~(size_t)255;
    return p;
  };
  unsigned short* xbf  = (unsigned short*)take((size_t)Bsz * D0 * 2);
  unsigned short* w0bf = (unsigned short*)take((size_t)D1 * D0 * 2);
  unsigned short* w1bf = (unsigned short*)take((size_t)D2 * D1 * 2);
  unsigned short* w2bf = (unsigned short*)take((size_t)D3 * D2 * 2);
  float* acts0         = (float*)take((size_t)Bsz * D1 * 4);
  float* acts1         = (float*)take((size_t)Bsz * D2 * 4);
  unsigned short* h1bf = (unsigned short*)take((size_t)Bsz * D1 * 2);
  unsigned short* h2bf = (unsigned short*)take((size_t)Bsz * D2 * 2);

  auto cvt = [&](const float* w, const float* m, unsigned short* o, int n) {
    cvt_masked_bf16<<<dim3((unsigned)((n + 255) / 256)), dim3(256), 0, stream>>>(w, m, o, n);
  };
  cvt(x,  nullptr, xbf,  Bsz * D0);
  cvt(W0, mask0,   w0bf, D1 * D0);
  cvt(W1, mask1,   w1bf, D2 * D1);
  cvt(W2, mask2,   w2bf, D3 * D2);

  // Layer 0: acts0 = x @ (W0*m0)^T + b0 ; fused h1bf = bf16(relu(acts0))
  gemm_bf16_wmma<<<dim3(D1 / 256, Bsz / 128), dim3(256), 0, stream>>>(
      xbf, w0bf, b0, acts0, h1bf, Bsz, D1, D0);

  // Layer 1: acts1 = relu(acts0) @ (W1*m1)^T + b1 + patch(t=0)  (relu after patch)
  gemm_bf16_wmma<<<dim3(D2 / 256, Bsz / 128), dim3(256), 0, stream>>>(
      h1bf, w1bf, b1, acts1, nullptr, Bsz, D2, D1);
  patch_kernel<<<dim3(Bsz / 256), dim3(256), 0, stream>>>(
      acts0, acts1, D2,
      hsrc + 0 * P, htgt + 0 * P * 4, lsmp + 0 * P * 20, ltgt + 0 * P,
      hW1 + 0 * P * 8, hb1 + 0 * P * 8, hW2 + 0 * P * 32, hb2 + 0 * P * 4,
      lW1 + 0 * P * 160, lb1 + 0 * P * 8, lW2 + 0 * P * 8, lb2 + 0 * P,
      Bsz, D1);
  relu_cvt_bf16<<<dim3((Bsz * D2 + 255) / 256), dim3(256), 0, stream>>>(
      acts1, h2bf, Bsz * D2);

  // Layer 2 (no relu): d_out = relu(acts1) @ (W2*m2)^T + b2 + patch(t=1)
  gemm_bf16_wmma<<<dim3(D3 / 256, Bsz / 128), dim3(256), 0, stream>>>(
      h2bf, w2bf, b2, (float*)d_out, nullptr, Bsz, D3, D2);
  patch_kernel<<<dim3(Bsz / 256), dim3(256), 0, stream>>>(
      acts1, (float*)d_out, D3,
      hsrc + 1 * P, htgt + 1 * P * 4, lsmp + 1 * P * 20, ltgt + 1 * P,
      hW1 + 1 * P * 8, hb1 + 1 * P * 8, hW2 + 1 * P * 32, hb2 + 1 * P * 4,
      lW1 + 1 * P * 160, lb1 + 1 * P * 8, lW2 + 1 * P * 8, lb2 + 1 * P,
      Bsz, D2);
}
